// SRPLayer_76613626626565
// MI455X (gfx1250) — matrix operand order repaired and checked by reading, not compiled, not run on hardware
//
#include <hip/hip_runtime.h>
#include <cstdint>

// ---------------------------------------------------------------------------
// SRP layer:  out = (Psi - rowmean(Psi)) @ x^T @ x
//   x   : [4096, 8192] f32
//   Psi : [2048, 8192] f32
//   out : [2048, 8192] f32
// Strategy: fp32 emulated on bf16 WMMA (hi/lo split, 3-term: hh + hl + lh),
// tiles staged with GLOBAL_LOAD_ASYNC_TO_LDS into double-buffered LDS.
// ---------------------------------------------------------------------------

typedef __attribute__((ext_vector_type(16))) __bf16 v16bf;
typedef __attribute__((ext_vector_type(8)))  float  v8f;
typedef __attribute__((ext_vector_type(4)))  int    v4i;

#define D_FEAT 4096
#define N_SAMP 8192
#define O_DIM  2048

// ---------------- bf16 helpers (round-to-nearest-even) ----------------------
__device__ __forceinline__ unsigned short f2bf(float f) {
    unsigned u = __float_as_uint(f);
    u += 0x7FFFu + ((u >> 16) & 1u);
    return (unsigned short)(u >> 16);
}
__device__ __forceinline__ float bf2f(unsigned short h) {
    return __uint_as_float(((unsigned)h) << 16);
}
__device__ __forceinline__ void split2(float f, unsigned short& h, unsigned short& l) {
    h = f2bf(f);
    l = f2bf(f - bf2f(h));
}

// ---------------- async global->LDS copy (16 bytes / lane) ------------------
#if __has_builtin(__builtin_amdgcn_global_load_async_to_lds_b128)
// Param types per hipcc diagnostic: b128 flavor takes int4 pointers,
// src in addrspace(1) (global), dst in addrspace(3) (LDS).
__device__ __forceinline__ void cp_async16(const unsigned short* g, unsigned short* s) {
    __builtin_amdgcn_global_load_async_to_lds_b128(
        (__attribute__((address_space(1))) v4i*)(uintptr_t)g,
        (__attribute__((address_space(3))) v4i*)(uintptr_t)s,
        0, 0);
}
#else
// Fall back to raw CDNA5 ISA: VDST = LDS byte address, VADDR = 64-bit global.
__device__ __forceinline__ void cp_async16(const unsigned short* g, unsigned short* s) {
    unsigned lds = (unsigned)(uintptr_t)s;   // generic addr low 32b == LDS offset
    asm volatile("global_load_async_to_lds_b128 %0, %1, off"
                 :: "v"(lds), "v"(g) : "memory");
}
#endif

__device__ __forceinline__ void wait_async0() {
#if __has_builtin(__builtin_amdgcn_s_wait_asynccnt)
    __builtin_amdgcn_s_wait_asynccnt(0);
#else
    asm volatile("s_wait_asynccnt 0x0" ::: "memory");
#endif
}

// ---------------- kernel 1: row means of Psi --------------------------------
__global__ __launch_bounds__(256) void mean_rows(const float* __restrict__ Psi,
                                                 float* __restrict__ mu, int n) {
    __shared__ float red[256];
    const int tid = threadIdx.x;
    const int row = blockIdx.x;
    const float* p = Psi + (size_t)row * n;
    float s = 0.0f;
    for (int i = tid; i < n; i += 256) s += p[i];
    red[tid] = s;
    __syncthreads();
    for (int off = 128; off > 0; off >>= 1) {
        if (tid < off) red[tid] += red[tid + off];
        __syncthreads();
    }
    if (tid == 0) mu[row] = red[0] * (1.0f / (float)n);
}

// ---------------- kernel 2: (optionally centered) hi/lo split ---------------
__global__ __launch_bounds__(256) void split_hl(const float* __restrict__ src,
                                                const float* __restrict__ mu,
                                                int rowLen4,
                                                unsigned short* __restrict__ h,
                                                unsigned short* __restrict__ l) {
    const int idx = blockIdx.x * 256 + threadIdx.x;
    float4 v = ((const float4*)src)[idx];
    float m = (mu != nullptr) ? mu[idx / rowLen4] : 0.0f;
    ushort4 hv, lv;
    split2(v.x - m, hv.x, lv.x);
    split2(v.y - m, hv.y, lv.y);
    split2(v.z - m, hv.z, lv.z);
    split2(v.w - m, hv.w, lv.w);
    ((ushort4*)h)[idx] = hv;
    ((ushort4*)l)[idx] = lv;
}

// ---------------- kernel 3: convert x -> bf16 pair + transposed bf16 pair ---
__global__ __launch_bounds__(256) void conv_xpose(const float* __restrict__ x,
                                                  unsigned short* __restrict__ xh,
                                                  unsigned short* __restrict__ xl,
                                                  unsigned short* __restrict__ xth,
                                                  unsigned short* __restrict__ xtl,
                                                  int R, int C) {
    __shared__ float t[32][33];
    const int tx = threadIdx.x, ty = threadIdx.y;
    const int c = blockIdx.x * 32 + tx;
    const int rbase = blockIdx.y * 32;
#pragma unroll
    for (int it = 0; it < 4; ++it) {
        const int r = rbase + ty + it * 8;
        const float f = x[(size_t)r * C + c];
        t[ty + it * 8][tx] = f;
        unsigned short h, l;
        split2(f, h, l);
        xh[(size_t)r * C + c] = h;
        xl[(size_t)r * C + c] = l;
    }
    __syncthreads();
    const int rr_base = blockIdx.x * 32;   // xT row = original column
    const int cc = rbase + tx;             // xT col = original row
#pragma unroll
    for (int it = 0; it < 4; ++it) {
        const int rr = rr_base + ty + it * 8;
        const float f = t[tx][ty + it * 8];
        unsigned short h, l;
        split2(f, h, l);
        xth[(size_t)rr * R + cc] = h;
        xtl[(size_t)rr * R + cc] = l;
    }
}

// ---------------- kernel 4: fp32-via-bf16x3 GEMM ----------------------------
// C[M,N] (f32) = (Ah+Al)[M,K] @ (Bh+Bl)[K,N], 3-term split.
// Block tile 128x128, BK=32, 256 threads = 8 wave32's (4M x 2N); each wave
// owns a 32x64 patch = 2x4 WMMA accumulators. Double-buffered LDS fed by
// async global->LDS loads; one barrier per k-step.
__device__ __forceinline__ v16bf ld_frag(const unsigned short* p0,
                                         const unsigned short* p1) {
    union { uint4 q[2]; v16bf v; } u;
    u.q[0] = *(const uint4*)p0;
    u.q[1] = *(const uint4*)p1;
    return u.v;
}

__global__ __launch_bounds__(256) void gemm_bf16x3(
        const unsigned short* __restrict__ Ah, const unsigned short* __restrict__ Al,
        const unsigned short* __restrict__ Bh, const unsigned short* __restrict__ Bl,
        float* __restrict__ C, int M, int N, int K) {
    constexpr int LDA = 40;    // 32 + 8 halfword pad: conflict-free b128 frag reads
    constexpr int LDB = 136;   // 128 + 8 halfword pad
    constexpr int ASZ = 128 * LDA;
    constexpr int BSZ = 32 * LDB;
    __shared__ __align__(16) unsigned short sAh[2 * ASZ];
    __shared__ __align__(16) unsigned short sAl[2 * ASZ];
    __shared__ __align__(16) unsigned short sBh[2 * BSZ];
    __shared__ __align__(16) unsigned short sBl[2 * BSZ];

    const int tid  = threadIdx.x;
    const int lane = tid & 31;
    const int wid  = tid >> 5;
    const int wm   = wid >> 1;      // 0..3 -> M offset wm*32
    const int wn   = wid & 1;       // 0..1 -> N offset wn*64
    const int m16  = lane & 15;
    const int g    = lane >> 4;     // half-wave group

    const int mBase = blockIdx.y * 128;
    const int nBase = blockIdx.x * 128;

    // staging coords: two 8-halfword (16 B) chunks per matrix per thread
    const int q0 = tid, q1 = tid + 256;
    const int ar0 = q0 >> 2, ac0 = (q0 & 3) * 8;
    const int ar1 = q1 >> 2, ac1 = (q1 & 3) * 8;
    const int br0 = q0 >> 4, bc0 = (q0 & 15) * 8;
    const int br1 = q1 >> 4, bc1 = (q1 & 15) * 8;
    const int la0 = ar0 * LDA + ac0, la1 = ar1 * LDA + ac1;
    const int lb0 = br0 * LDB + bc0, lb1 = br1 * LDB + bc1;

    const int KT = K >> 5;

    auto issueTile = [&](int kt, int sel) {
        const size_t a0 = (size_t)(mBase + ar0) * K + kt * 32 + ac0;
        const size_t a1 = (size_t)(mBase + ar1) * K + kt * 32 + ac1;
        const size_t b0 = (size_t)(kt * 32 + br0) * N + nBase + bc0;
        const size_t b1 = (size_t)(kt * 32 + br1) * N + nBase + bc1;
        const int ao = sel * ASZ, bo = sel * BSZ;
        cp_async16(Ah + a0, &sAh[ao + la0]);
        cp_async16(Ah + a1, &sAh[ao + la1]);
        cp_async16(Al + a0, &sAl[ao + la0]);
        cp_async16(Al + a1, &sAl[ao + la1]);
        cp_async16(Bh + b0, &sBh[bo + lb0]);
        cp_async16(Bh + b1, &sBh[bo + lb1]);
        cp_async16(Bl + b0, &sBl[bo + lb0]);
        cp_async16(Bl + b1, &sBl[bo + lb1]);
    };

    v8f acc[2][4];
    const v8f vzero = {};
#pragma unroll
    for (int i = 0; i < 2; ++i)
#pragma unroll
        for (int j = 0; j < 4; ++j) acc[i][j] = vzero;

    issueTile(0, 0);
    wait_async0();
    __syncthreads();

    for (int kt = 0; kt < KT; ++kt) {
        const int cur = kt & 1;
        if (kt + 1 < KT) issueTile(kt + 1, cur ^ 1);     // async into other buffer
        if (kt + 2 < KT) {  // GL2 hint two tiles ahead -> global_prefetch_b8
            __builtin_prefetch(Ah + (size_t)(mBase + ar0) * K + (kt + 2) * 32 + ac0, 0, 1);
            __builtin_prefetch(Bh + (size_t)((kt + 2) * 32 + br0) * N + nBase + bc0, 0, 1);
        }

        const int ao = cur * ASZ, bo = cur * BSZ;

        // A fragments: lane m16 holds row m; K chunks per ISA 16-bit A layout.
        v16bf ah[2], al[2];
#pragma unroll
        for (int i = 0; i < 2; ++i) {
            const int arow = ao + (wm * 32 + i * 16 + m16) * LDA;
            ah[i] = ld_frag(&sAh[arow + g * 8], &sAh[arow + 16 + g * 8]);
            al[i] = ld_frag(&sAl[arow + g * 8], &sAl[arow + 16 + g * 8]);
        }
        // B fragments: lane holds N column m16; K chunks match the A layout.
        v16bf bh[4], bl[4];
        const int krow = bo + m16;
#pragma unroll
        for (int j = 0; j < 4; ++j) {
            const int bcol = krow + wn * 64 + j * 16;
            union { unsigned short u[16]; v16bf v; } th, tl;
#pragma unroll
            for (int s = 0; s < 16; ++s) {
                const int k = (s < 8) ? (g * 8 + s) : (16 + g * 8 + (s - 8));
                th.u[s] = sBh[bcol + k * LDB];
                tl.u[s] = sBl[bcol + k * LDB];
            }
            bh[j] = th.v;
            bl[j] = tl.v;
        }

#pragma unroll
        for (int i = 0; i < 2; ++i)
#pragma unroll
            for (int j = 0; j < 4; ++j) {
                acc[i][j] = __builtin_amdgcn_wmma_f32_16x16x32_bf16(
                    false, ah[i], false, bh[j], (short)0, acc[i][j], false, false);
                acc[i][j] = __builtin_amdgcn_wmma_f32_16x16x32_bf16(
                    false, ah[i], false, bl[j], (short)0, acc[i][j], false, false);
                acc[i][j] = __builtin_amdgcn_wmma_f32_16x16x32_bf16(
                    false, al[i], false, bh[j], (short)0, acc[i][j], false, false);
            }

        wait_async0();       // our tile k+1 loads are in LDS
        __syncthreads();     // everyone done computing on cur + everyone's loads done
    }

    // Epilogue: C/D layout — VGPR r holds M = r + 8*g, lane&15 holds N.
#pragma unroll
    for (int i = 0; i < 2; ++i) {
#pragma unroll
        for (int j = 0; j < 4; ++j) {
            const int col = nBase + wn * 64 + j * 16 + m16;
#pragma unroll
            for (int r = 0; r < 8; ++r) {
                const int row = mBase + wm * 32 + i * 16 + r + 8 * g;
                C[(size_t)row * N + col] = acc[i][j][r];
            }
        }
    }
}

// ---------------------------------------------------------------------------
extern "C" void kernel_launch(void* const* d_in, const int* in_sizes, int n_in,
                              void* d_out, int out_size, void* d_ws, size_t ws_size,
                              hipStream_t stream) {
    const float* x   = (const float*)d_in[0];   // [4096, 8192]
    const float* Psi = (const float*)d_in[1];   // [2048, 8192]
    float* out = (float*)d_out;                 // [2048, 8192]
    char* ws = (char*)d_ws;

    const size_t SZ_MU   = 8192;                                   // 2048 f32
    const size_t SZ_PC   = (size_t)O_DIM * N_SAMP * 2;             // 32 MiB each
    const size_t SZ_X    = (size_t)D_FEAT * N_SAMP * 2;            // 64 MiB each
    const size_t SZ_TMP  = (size_t)O_DIM * D_FEAT * 4;             // 32 MiB
    const size_t SZ_TMPH = (size_t)O_DIM * D_FEAT * 2;             // 16 MiB each

    size_t off = 0;
    float*          mu   = (float*)(ws + off);          off += SZ_MU;
    unsigned short* Pch  = (unsigned short*)(ws + off); off += SZ_PC;
    unsigned short* Pcl  = (unsigned short*)(ws + off); off += SZ_PC;
    unsigned short* xh   = (unsigned short*)(ws + off); off += SZ_X;
    unsigned short* xl   = (unsigned short*)(ws + off); off += SZ_X;
    unsigned short* xth  = (unsigned short*)(ws + off); off += SZ_X;
    unsigned short* xtl  = (unsigned short*)(ws + off); off += SZ_X;
    float*          tmp  = (float*)(ws + off);          off += SZ_TMP;
    unsigned short* tmph = (unsigned short*)(ws + off); off += SZ_TMPH;
    unsigned short* tmpl = (unsigned short*)(ws + off); off += SZ_TMPH;

    // 1. row means of Psi
    mean_rows<<<O_DIM, 256, 0, stream>>>(Psi, mu, N_SAMP);

    // 2. Psi_c hi/lo split (centered)
    split_hl<<<(O_DIM * N_SAMP / 4) / 256, 256, 0, stream>>>(
        Psi, mu, N_SAMP / 4, Pch, Pcl);

    // 3. x -> bf16 pair (row-major) + transposed bf16 pair
    conv_xpose<<<dim3(N_SAMP / 32, D_FEAT / 32), dim3(32, 8), 0, stream>>>(
        x, xh, xl, xth, xtl, D_FEAT, N_SAMP);

    // 4. GEMM1: tmp[o,d] = Psi_c @ x^T   (A = Pc [o,n], B = xT [n,d])
    gemm_bf16x3<<<dim3(D_FEAT / 128, O_DIM / 128), 256, 0, stream>>>(
        Pch, Pcl, xth, xtl, tmp, O_DIM, D_FEAT, N_SAMP);

    // 5. tmp hi/lo split
    split_hl<<<(O_DIM * D_FEAT / 4) / 256, 256, 0, stream>>>(
        tmp, nullptr, D_FEAT / 4, tmph, tmpl);

    // 6. GEMM2: out[o,n] = tmp @ x      (A = tmp [o,d], B = x [d,n])
    gemm_bf16x3<<<dim3(N_SAMP / 128, O_DIM / 128), 256, 0, stream>>>(
        tmph, tmpl, xh, xl, out, O_DIM, N_SAMP, D_FEAT);
}
